// AttentionOperation_61572651156145
// MI455X (gfx1250) — matrix operand order repaired and checked by reading, not compiled, not run on hardware
//
#include <hip/hip_runtime.h>
#include <math.h>

typedef __bf16 bf16;
typedef __attribute__((ext_vector_type(16))) __bf16 v16bf;
typedef __attribute__((ext_vector_type(8)))  float  v8f;

#define N_B 4
#define N_H 8
#define N_D 64
#define N_L 1024
#define N_C 64
#define CHV (N_H * N_C)   // 512 value channels

__device__ inline v8f vzero8() {
  v8f z;
#pragma unroll
  for (int i = 0; i < 8; ++i) z[i] = 0.0f;
  return z;
}

// ---------------- pass 0: fp32 -> bf16 conversion (+ transposed K copy) ----------------
__global__ void cvt_bf16_kernel(const float* __restrict__ q, const float* __restrict__ k,
                                const float* __restrict__ v,
                                bf16* __restrict__ qb, bf16* __restrict__ kb,
                                bf16* __restrict__ vb, bf16* __restrict__ kbT, int n) {
  int i = blockIdx.x * blockDim.x + threadIdx.x;
  if (i < n) {
    qb[i] = (bf16)q[i];
    bf16 kv = (bf16)k[i];
    kb[i] = kv;
    vb[i] = (bf16)v[i];
    // kbT layout: [b,h,m,d] so WMMA B-fragments of K are d-contiguous per lane
    int m  = i % N_L;
    int dd = (i / N_L) % N_D;
    int bh = i / (N_L * N_D);
    kbT[((size_t)bh * N_L + m) * N_D + dd] = kv;
  }
}

// ---------------- zero the atomically-accumulated head stats ----------------
__global__ void zero_stats_kernel(float* hsum, float* hsumsq) {
  int i = threadIdx.x;
  if (i < N_H) { hsum[i] = 0.0f; hsumsq[i] = 0.0f; }
}

// ---------------- Sum(sim) per head via factored row sums ----------------
// sum_{l,m} sim = sum_d (sum_l q[d,l]) * (sum_m k[d,m])
__global__ void head_sums_kernel(const bf16* __restrict__ qb, const bf16* __restrict__ kb,
                                 float* __restrict__ hsum) {
  int bh = blockIdx.x;            // N_B*N_H blocks
  int h  = bh % N_H;
  int d  = threadIdx.x;           // 64 threads
  const bf16* qrow = qb + ((size_t)bh * N_D + d) * N_L;
  const bf16* krow = kb + ((size_t)bh * N_D + d) * N_L;
  float sq = 0.0f, sk = 0.0f;
  for (int l = 0; l < N_L; ++l) { sq += (float)qrow[l]; sk += (float)krow[l]; }
  __shared__ float red[64];
  red[d] = sq * sk;
  __syncthreads();
  for (int s = 32; s > 0; s >>= 1) { if (d < s) red[d] += red[d + s]; __syncthreads(); }
  if (d == 0) atomicAdd(&hsum[h], red[0]);
}

// ---------------- Sum(sim^2) per head via Gram matrices (WMMA) ----------------
// ||Q^T K||_F^2 = <Q Q^T, K K^T>_F  with Q,K of shape D x L (D=64)
__global__ void head_sumsq_kernel(const bf16* __restrict__ qb, const bf16* __restrict__ kb,
                                  float* __restrict__ hsumsq) {
  int bh   = blockIdx.x;              // N_B*N_H blocks, 512 threads (16 waves)
  int h    = bh % N_H;
  int wave = threadIdx.x >> 5;
  int lane = threadIdx.x & 31;
  int hl   = lane >> 4;
  int mrow = lane & 15;
  int dt   = wave >> 2;               // row tile of the 64x64 Gram
  int et   = wave & 3;                // col tile
  const bf16* qbase = qb + (size_t)bh * N_D * N_L;
  const bf16* kbase = kb + (size_t)bh * N_D * N_L;

  v8f gq = vzero8();
  v8f gk = vzero8();
  for (int kc = 0; kc < N_L / 32; ++kc) {
    v16bf aq, bq, ak, bk;
#pragma unroll
    for (int v = 0; v < 8; ++v) {
      int ka = (v < 4) ? (2 * v + 8 * hl) : (16 + 2 * (v - 4) + 8 * hl); // A K-pair base
      int kb2 = ka;                                                       // B K-pair base
      size_t la = (size_t)(dt * 16 + mrow) * N_L + kc * 32 + ka;
      size_t lb = (size_t)(et * 16 + mrow) * N_L + kc * 32 + kb2;
      aq[2 * v] = qbase[la];     aq[2 * v + 1] = qbase[la + 1];
      bq[2 * v] = qbase[lb];     bq[2 * v + 1] = qbase[lb + 1];
      ak[2 * v] = kbase[la];     ak[2 * v + 1] = kbase[la + 1];
      bk[2 * v] = kbase[lb];     bk[2 * v + 1] = kbase[lb + 1];
    }
    gq = __builtin_amdgcn_wmma_f32_16x16x32_bf16(false, aq, false, bq, (short)0, gq, false, false);
    gk = __builtin_amdgcn_wmma_f32_16x16x32_bf16(false, ak, false, bk, (short)0, gk, false, false);
  }
  float part = 0.0f;
#pragma unroll
  for (int r = 0; r < 8; ++r) part += gq[r] * gk[r];
  for (int off = 16; off > 0; off >>= 1) part += __shfl_xor(part, off, 32);
  __shared__ float wred[16];
  if (lane == 0) wred[wave] = part;
  __syncthreads();
  if (threadIdx.x == 0) {
    float t = 0.0f;
    for (int i = 0; i < 16; ++i) t += wred[i];
    atomicAdd(&hsumsq[h], t);
  }
}

// ---------------- fused QK^T -> BN affine -> softmax -> AV ----------------
// One wave per 16-row stripe of sim; whole 16x1024 stripe lives in LDS.
__global__ void attn_fused_kernel(const bf16* __restrict__ qb, const bf16* __restrict__ kbT,
                                  const bf16* __restrict__ vb,
                                  const float* __restrict__ hsum, const float* __restrict__ hsumsq,
                                  const float* __restrict__ wsim, const float* __restrict__ bsim,
                                  float* __restrict__ rvbuf) {
  extern __shared__ char smem[];
  float* simf = (float*)smem;                        // 16*1024 f32  (64 KB)
  bf16*  expb = (bf16*)(smem + 16 * N_L * 4);        // 16*1024 bf16 (32 KB)
  float* dinv = (float*)(smem + 16 * N_L * 6);       // 16 f32 reciprocal denominators

  int lt = blockIdx.x, h = blockIdx.y, b = blockIdx.z;
  int bh = b * N_H + h;
  int lane = threadIdx.x & 31;
  int hl = lane >> 4, mrow = lane & 15;
  int l0 = lt * 16;

  const float cnt = (float)N_B * (float)N_L * (float)N_L;
  float mean = hsum[h] / cnt;
  float var  = hsumsq[h] / cnt - mean * mean;
  float rstd = rsqrtf(var + 1e-3f);
  float sc = rstd * wsim[h];
  float sh = bsim[h] - mean * sc;

  const bf16* qb_  = qb  + (size_t)bh * N_D * N_L;
  const bf16* kbT_ = kbT + (size_t)bh * N_L * N_D;   // [m][d]
  const bf16* vb_  = vb  + (size_t)bh * N_C * N_L;

  // A fragments of Q^T (M=l-local, K=d): two 32-wide d chunks, reused for all m tiles
  v16bf aq[2];
#pragma unroll
  for (int c = 0; c < 2; ++c) {
#pragma unroll
    for (int v = 0; v < 8; ++v) {
      int ka = (v < 4) ? (2 * v + 8 * hl) : (16 + 2 * (v - 4) + 8 * hl);
      int d = c * 32 + ka;
      aq[c][2 * v]     = qb_[(size_t)d * N_L + l0 + mrow];
      aq[c][2 * v + 1] = qb_[(size_t)(d + 1) * N_L + l0 + mrow];
    }
  }

  // sim stripe: 64 m-tiles of 16, K over d in two chunks
  for (int mt = 0; mt < N_L / 16; ++mt) {
    v8f c8 = vzero8();
    int m = mt * 16 + mrow;
#pragma unroll
    for (int c = 0; c < 2; ++c) {
      v16bf bk;
#pragma unroll
      for (int v = 0; v < 8; ++v) {
        int d = c * 32 + ((v < 4) ? (2 * v + 8 * hl) : (16 + 2 * (v - 4) + 8 * hl));  // table K order, matches A
        bk[2 * v]     = kbT_[(size_t)m * N_D + d];
        bk[2 * v + 1] = kbT_[(size_t)m * N_D + d + 1];
      }
      c8 = __builtin_amdgcn_wmma_f32_16x16x32_bf16(false, aq[c], false, bk, (short)0, c8, false, false);
    }
#pragma unroll
    for (int r = 0; r < 8; ++r) {
      int M = r + 8 * hl;
      simf[M * N_L + mt * 16 + mrow] = c8[r] * sc + sh;   // BN affine folded in
    }
  }
  __syncthreads();

  // softmax: lane i (<16) owns row i of the stripe; store unnormalized exp as bf16
  if (lane < 16) {
    int row = lane;
    const float4* rp = (const float4*)&simf[row * N_L];
    float mx = -1e30f;
    for (int i = 0; i < N_L / 4; ++i) {
      float4 t = rp[i];
      mx = fmaxf(mx, fmaxf(fmaxf(t.x, t.y), fmaxf(t.z, t.w)));
    }
    float s = 0.0f;
    for (int i = 0; i < N_L / 4; ++i) {
      float4 t = rp[i];
      float e0 = __expf(t.x - mx), e1 = __expf(t.y - mx);
      float e2 = __expf(t.z - mx), e3 = __expf(t.w - mx);
      s += e0 + e1 + e2 + e3;
      bf16* ep = &expb[row * N_L + i * 4];
      ep[0] = (bf16)e0; ep[1] = (bf16)e1; ep[2] = (bf16)e2; ep[3] = (bf16)e3;
    }
    dinv[row] = 1.0f / s;
  }
  __syncthreads();

  // AV: D tile (l x d), A = exp stripe (K=m), B = V^T (K=m, N=d); 4 d-tiles
  v8f acc[4] = { vzero8(), vzero8(), vzero8(), vzero8() };
  for (int kc = 0; kc < N_L / 32; ++kc) {
    v16bf a;
#pragma unroll
    for (int v = 0; v < 8; ++v) {
      int ka = (v < 4) ? (2 * v + 8 * hl) : (16 + 2 * (v - 4) + 8 * hl);
      int m = kc * 32 + ka;
      a[2 * v]     = expb[mrow * N_L + m];
      a[2 * v + 1] = expb[mrow * N_L + m + 1];
    }
#pragma unroll
    for (int n = 0; n < 4; ++n) {
      v16bf bv;
#pragma unroll
      for (int v = 0; v < 8; ++v) {
        int m = kc * 32 + ((v < 4) ? (2 * v + 8 * hl) : (16 + 2 * (v - 4) + 8 * hl));  // table K order, matches A
        int d = n * 16 + mrow;
        bv[2 * v]     = vb_[(size_t)d * N_L + m];
        bv[2 * v + 1] = vb_[(size_t)d * N_L + m + 1];
      }
      acc[n] = __builtin_amdgcn_wmma_f32_16x16x32_bf16(false, a, false, bv, (short)0, acc[n], false, false);
    }
  }

  // normalize rows by softmax denominator and write rv in (N, H*C, L) layout
#pragma unroll
  for (int n = 0; n < 4; ++n) {
#pragma unroll
    for (int r = 0; r < 8; ++r) {
      int M = r + 8 * hl;
      int l = l0 + M;
      int d = n * 16 + mrow;
      rvbuf[((size_t)b * CHV + h * N_C + d) * N_L + l] = acc[n][r] * dinv[M];
    }
  }
}

// ---------------- per-channel stats of rv ----------------
__global__ void chan_stats_kernel(const float* __restrict__ rvbuf,
                                  float* __restrict__ csum, float* __restrict__ csumsq) {
  int ch = blockIdx.x;        // CHV blocks
  int t = threadIdx.x;        // 256 threads
  float s = 0.0f, ss = 0.0f;
  for (int b = 0; b < N_B; ++b) {
    const float* p = rvbuf + ((size_t)b * CHV + ch) * N_L;
    for (int l = t; l < N_L; l += 256) { float x = p[l]; s += x; ss += x * x; }
  }
  __shared__ float rs[256], rss[256];
  rs[t] = s; rss[t] = ss;
  __syncthreads();
  for (int o = 128; o > 0; o >>= 1) {
    if (t < o) { rs[t] += rs[t + o]; rss[t] += rss[t + o]; }
    __syncthreads();
  }
  if (t == 0) { csum[ch] = rs[0]; csumsq[ch] = rss[0]; }
}

// ---------------- BN + exact GELU epilogue ----------------
__global__ void bn_gelu_kernel(const float* __restrict__ rvbuf,
                               const float* __restrict__ csum, const float* __restrict__ csumsq,
                               const float* __restrict__ wv, const float* __restrict__ bv,
                               float* __restrict__ out) {
  int i = blockIdx.x * blockDim.x + threadIdx.x;
  if (i >= N_B * CHV * N_L) return;
  int ch = (i / N_L) % CHV;
  const float cnt = (float)(N_B * N_L);
  float mean = csum[ch] / cnt;
  float var  = csumsq[ch] / cnt - mean * mean;
  float rstd = rsqrtf(var + 1e-3f);
  float x = (rvbuf[i] - mean) * rstd * wv[ch] + bv[ch];
  out[i] = 0.5f * x * (1.0f + erff(x * 0.70710678118654752f));
}

extern "C" void kernel_launch(void* const* d_in, const int* in_sizes, int n_in,
                              void* d_out, int out_size, void* d_ws, size_t ws_size,
                              hipStream_t stream) {
  const float* q    = (const float*)d_in[0];
  const float* k    = (const float*)d_in[1];
  const float* v    = (const float*)d_in[2];
  const float* wsim = (const float*)d_in[3];
  const float* bsim = (const float*)d_in[4];
  const float* wv   = (const float*)d_in[5];
  const float* bv   = (const float*)d_in[6];

  const size_t nQ = (size_t)N_B * N_H * N_D * N_L;   // 2M elements (q, k, v, rv all equal)
  char* ws = (char*)d_ws;
  bf16* qb     = (bf16*)ws;  ws += nQ * sizeof(bf16);
  bf16* kb     = (bf16*)ws;  ws += nQ * sizeof(bf16);
  bf16* vb     = (bf16*)ws;  ws += nQ * sizeof(bf16);
  bf16* kbT    = (bf16*)ws;  ws += nQ * sizeof(bf16);
  float* rvbuf = (float*)ws; ws += nQ * sizeof(float);
  float* hsum  = (float*)ws; ws += N_H * sizeof(float);
  float* hsumsq= (float*)ws; ws += N_H * sizeof(float);
  float* csum  = (float*)ws; ws += CHV * sizeof(float);
  float* csumsq= (float*)ws; ws += CHV * sizeof(float);

  int n = (int)nQ;
  cvt_bf16_kernel<<<(n + 255) / 256, 256, 0, stream>>>(q, k, v, qb, kb, vb, kbT, n);
  zero_stats_kernel<<<1, 32, 0, stream>>>(hsum, hsumsq);
  head_sums_kernel<<<N_B * N_H, 64, 0, stream>>>(qb, kb, hsum);
  head_sumsq_kernel<<<N_B * N_H, 512, 0, stream>>>(qb, kb, hsumsq);

  size_t shmem = (size_t)16 * N_L * 4 + (size_t)16 * N_L * 2 + 16 * 4;  // ~96 KB
  attn_fused_kernel<<<dim3(N_L / 16, N_H, N_B), 32, shmem, stream>>>(
      qb, kbT, vb, hsum, hsumsq, wsim, bsim, rvbuf);

  chan_stats_kernel<<<CHV, 256, 0, stream>>>(rvbuf, csum, csumsq);
  bn_gelu_kernel<<<(n + 255) / 256, 256, 0, stream>>>(rvbuf, csum, csumsq, wv, bv, (float*)d_out);
}
